// SEP_U_84988812853303
// MI455X (gfx1250) — matrix operand order repaired and checked by reading, not compiled, not run on hardware
//
#include <hip/hip_runtime.h>
#include <hip/hip_bf16.h>
#include <cstddef>

// SEP-U hierarchical GCN for MI455X (gfx1250, wave32).
// Dense math: f16 WMMA (v_wmma_f32_16x16x32_f16), f32 accumulate.
//   - weights pre-converted to f16 and transposed (Bt[n][k]) so a lane's B
//     fragment is two 16B loads; N padded to 16 so no guards.
//   - all M dims are multiples of 16 -> no M guards in the hot loop.
//   - K templated (128/256) -> fully unrolled chunk loop, clustered b128 loads.
// Edge/pool aggregation: wave-per-row float4 gathers + f32 atomics (L2-resident).

typedef __attribute__((ext_vector_type(16))) _Float16 v16h;
typedef __attribute__((ext_vector_type(8)))  float    v8f;

#define HDIM 128

// ---------------------------------------------------------------- GEMM (WMMA)
// Y[M,N] = (A @ B) + bias.  A rows may be the concat of two K0/K-K0 sources
// (per-chunk select; K0 multiple of 32) and may be gathered through aidx.
// Bt is [Np][K] f16 (transposed weight), zero-padded rows for n >= N.
template <int K>
__global__ __launch_bounds__(256) void wmma_gemm_k(
    const float* __restrict__ A0, const float* __restrict__ A1, int K0,
    const int* __restrict__ aidx,
    const _Float16* __restrict__ Bt, const float* __restrict__ bias,
    float* __restrict__ Y, int M, int N, int Np)
{
  const int tilesN = Np >> 4;
  const int tilesM = M >> 4;                 // M is a multiple of 16
  const int total  = tilesM * tilesN;
  const int wave   = blockIdx.x * (blockDim.x >> 5) + (threadIdx.x >> 5);
  if (wave >= total) return;                 // wave-uniform: EXEC stays full
  const int tm = wave / tilesN, tn = wave - tm * tilesN;
  const int m0 = tm << 4, n0 = tn << 4;
  const int lane = threadIdx.x & 31;
  const int half = lane >> 4;
  const int lm   = lane & 15;

  const int gmA  = m0 + lm;                  // A row this lane owns
  const int rowA = aidx ? aidx[gmA] : gmA;
  const int gn   = n0 + lm;                  // B/C column this lane owns
  const _Float16* bp = Bt + (size_t)gn * K;

  v8f acc = {};
#pragma unroll
  for (int kc = 0; kc < K; kc += 32) {
    // chunk lies fully inside A0 or A1 (K0 multiple of 32)
    const float* arow = (kc < K0) ? (A0 + (size_t)rowA * K0 + kc)
                                  : (A1 + (size_t)rowA * (K - K0) + (kc - K0));
    // A 16x32 f16 fragment: two contiguous 8-f32 runs per lane
    const float4 f0 = *(const float4*)(arow + half * 8);
    const float4 f1 = *(const float4*)(arow + half * 8 + 4);
    const float4 f2 = *(const float4*)(arow + 16 + half * 8);
    const float4 f3 = *(const float4*)(arow + 16 + half * 8 + 4);
    v16h a;
    a[0]  = (_Float16)f0.x; a[1]  = (_Float16)f0.y;
    a[2]  = (_Float16)f0.z; a[3]  = (_Float16)f0.w;
    a[4]  = (_Float16)f1.x; a[5]  = (_Float16)f1.y;
    a[6]  = (_Float16)f1.z; a[7]  = (_Float16)f1.w;
    a[8]  = (_Float16)f2.x; a[9]  = (_Float16)f2.y;
    a[10] = (_Float16)f2.z; a[11] = (_Float16)f2.w;
    a[12] = (_Float16)f3.x; a[13] = (_Float16)f3.y;
    a[14] = (_Float16)f3.z; a[15] = (_Float16)f3.w;
    // B 32x16 f16 fragment: two 8-f16 (16B) runs at k = half*8 and 16 + half*8 from transposed weight
    typedef __attribute__((ext_vector_type(8))) _Float16 v8h_;
    const v8h_ b0 = *(const v8h_*)(bp + kc + half * 8);
    const v8h_ b1 = *(const v8h_*)(bp + kc + 16 + half * 8);
    const v16h b = __builtin_shufflevector(b0, b1, 0, 1, 2, 3, 4, 5, 6, 7, 8, 9, 10, 11, 12, 13, 14, 15);
    acc = __builtin_amdgcn_wmma_f32_16x16x32_f16(false, a, false, b,
                                                 (short)0, acc, false, false);
  }
  const bool nvalid = gn < N;
  const float bb = (bias != nullptr && nvalid) ? bias[gn] : 0.f;
#pragma unroll
  for (int v = 0; v < 8; ++v) {              // C/D: M = v + 8*half, N = gn
    const int gm = m0 + v + (half << 3);
    if (nvalid) Y[(size_t)gm * N + gn] = acc[v] + bb;
  }
}

// W[K,N] f32 -> Bt[Np,K] f16 transposed, zero-padded for n >= N
__global__ void convert_wT_k(const float* __restrict__ W, _Float16* __restrict__ Bt,
                             int K, int N, int Np) {
  int idx = blockIdx.x * blockDim.x + threadIdx.x;
  if (idx >= Np * K) return;
  int n = idx / K, k = idx - n * K;
  float v = (n < N) ? W[(size_t)k * N + n] : 0.f;
  Bt[idx] = (_Float16)v;
}

// ---------------------------------------------------------------- elementwise
__global__ void fill_k(float* p, float v, int n) {
  int i = blockIdx.x * blockDim.x + threadIdx.x;
  if (i < n) p[i] = v;
}

__global__ void deg_count_k(const int* __restrict__ cols, int e, float* deg) {
  int i = blockIdx.x * blockDim.x + threadIdx.x;
  if (i < e) atomicAdd(&deg[cols[i]], 1.0f);
}

__global__ void rsqrt_k(float* d, int n) {
  int i = blockIdx.x * blockDim.x + threadIdx.x;
  if (i < n) d[i] = rsqrtf(d[i]);
}

__global__ void bias_relu_k(float* y, const float* __restrict__ b, int total) {
  int i = blockIdx.x * blockDim.x + threadIdx.x;
  if (i >= total) return;
  float v = y[i] + b[i & (HDIM - 1)];
  y[i] = v > 0.f ? v : 0.f;
}

// relu -> BN(eval) -> relu, in place (GEMM already added linear bias)
__global__ void bn_relu_k(float* y, const float* __restrict__ g,
                          const float* __restrict__ beta,
                          const float* __restrict__ mean,
                          const float* __restrict__ var, int total) {
  int i = blockIdx.x * blockDim.x + threadIdx.x;
  if (i >= total) return;
  int c = i & (HDIM - 1);
  float t = y[i]; t = t > 0.f ? t : 0.f;
  t = (t - mean[c]) * (g[c] * rsqrtf(var[c] + 1e-5f)) + beta[c];
  y[i] = t > 0.f ? t : 0.f;
}

// ---------------------------------------------------------------- aggregation
// out[i,:] = dis[i]^2 * xw[i,:]   (self-loop init; full write)
__global__ __launch_bounds__(256) void agg_self128(
    const float* __restrict__ xw, const float* __restrict__ dis,
    float* __restrict__ out, int n) {
  int row = blockIdx.x * 8 + (threadIdx.x >> 5);
  if (row >= n) return;
  int lane = threadIdx.x & 31;
  float w = dis[row]; w *= w;
  float4 v = *(const float4*)(xw + (size_t)row * HDIM + lane * 4);
  float4 r = make_float4(w * v.x, w * v.y, w * v.z, w * v.w);
  *(float4*)(out + (size_t)row * HDIM + lane * 4) = r;
}

// out[col,:] += dis[row]*dis[col] * xw[row,:]   (one wave per edge)
__global__ __launch_bounds__(256) void agg_edge128(
    const int* __restrict__ rows, const int* __restrict__ cols,
    const float* __restrict__ dis, const float* __restrict__ xw,
    float* __restrict__ out, int E) {
  int e = blockIdx.x * 8 + (threadIdx.x >> 5);
  if (e >= E) return;
  int lane = threadIdx.x & 31;
  int r = rows[e], c = cols[e];
  float w = dis[r] * dis[c];
  float4 v = *(const float4*)(xw + (size_t)r * HDIM + lane * 4);
  float* o = out + (size_t)c * HDIM + lane * 4;
  atomicAdd(o + 0, w * v.x); atomicAdd(o + 1, w * v.y);
  atomicAdd(o + 2, w * v.z); atomicAdd(o + 3, w * v.w);
}

// p[parent[i],:] += h[i,:]  (segment-sum pooling; one wave per child row)
__global__ __launch_bounds__(256) void scatter_parent128(
    const float* __restrict__ h, const int* __restrict__ parent,
    float* __restrict__ p, int n) {
  int row = blockIdx.x * 8 + (threadIdx.x >> 5);
  if (row >= n) return;
  int lane = threadIdx.x & 31;
  int pr = parent[row];
  float4 v = *(const float4*)(h + (size_t)row * HDIM + lane * 4);
  float* o = p + (size_t)pr * HDIM + lane * 4;
  atomicAdd(o + 0, v.x); atomicAdd(o + 1, v.y);
  atomicAdd(o + 2, v.z); atomicAdd(o + 3, v.w);
}

// narrow (C=10) variants for the final GCN
__global__ void agg_self_c(const float* __restrict__ xw, const float* __restrict__ dis,
                           float* __restrict__ out, int n, int Cw) {
  int idx = blockIdx.x * blockDim.x + threadIdx.x;
  if (idx >= n * Cw) return;
  int i = idx / Cw;
  float w = dis[i];
  out[idx] = w * w * xw[idx];
}

__global__ void agg_edge_c(const int* __restrict__ rows, const int* __restrict__ cols,
                           const float* __restrict__ dis, const float* __restrict__ xw,
                           float* __restrict__ out, int E, int Cw) {
  int idx = blockIdx.x * blockDim.x + threadIdx.x;
  if (idx >= E * Cw) return;
  int e = idx / Cw, c = idx - e * Cw;
  int r = rows[e], t = cols[e];
  atomicAdd(&out[(size_t)t * Cw + c], dis[r] * dis[t] * xw[(size_t)r * Cw + c]);
}

__global__ void bias_add_c(float* y, const float* __restrict__ b, int total, int Cw) {
  int i = blockIdx.x * blockDim.x + threadIdx.x;
  if (i < total) y[i] += b[i % Cw];
}

__global__ void log_softmax_k(float* y, int n, int Cw) {
  int i = blockIdx.x * blockDim.x + threadIdx.x;
  if (i >= n) return;
  float* r = y + (size_t)i * Cw;
  float m = r[0];
  for (int c = 1; c < Cw; ++c) m = fmaxf(m, r[c]);
  float s = 0.f;
  for (int c = 0; c < Cw; ++c) s += __expf(r[c] - m);
  float lse = m + __logf(s);
  for (int c = 0; c < Cw; ++c) r[c] -= lse;
}

// ---------------------------------------------------------------- host driver
extern "C" void kernel_launch(void* const* d_in, const int* in_sizes, int n_in,
                              void* d_out, int out_size, void* d_ws, size_t ws_size,
                              hipStream_t stream) {
  const int N0 = 100000, N1 = 20000, N2 = 4000, CC = 10;
  const int E0 = in_sizes[1] / 2, E1 = in_sizes[2] / 2, E2 = in_sizes[3] / 2;
  (void)n_in; (void)out_size; (void)ws_size;

  const float* x    = (const float*)d_in[0];
  const int*   ei0  = (const int*)d_in[1];
  const int*   ei1  = (const int*)d_in[2];
  const int*   ei2  = (const int*)d_in[3];
  const int*   par1 = (const int*)d_in[4];
  const int*   par2 = (const int*)d_in[5];
  const float* W0 = (const float*)d_in[6];  const float* b0 = (const float*)d_in[7];
  const float* W1 = (const float*)d_in[8];  const float* b1 = (const float*)d_in[9];
  const float* W2 = (const float*)d_in[10]; const float* b2 = (const float*)d_in[11];
  const float* W3 = (const float*)d_in[12]; const float* b3 = (const float*)d_in[13];
  const float* W4 = (const float*)d_in[14]; const float* b4 = (const float*)d_in[15];
  const float* pW = (const float*)d_in[16]; const float* pb = (const float*)d_in[17];
  const float* pg = (const float*)d_in[18]; const float* pbt = (const float*)d_in[19];
  const float* pmn = (const float*)d_in[20]; const float* pvr = (const float*)d_in[21];
  float* out = (float*)d_out;

  // ---- workspace arena: f32 buffers (~155 MB), then f16 weight arena
  float* w = (float*)d_ws;
  size_t off = 0;
  auto alloc = [&](size_t n) { float* p = w + off; off += n; return p; };
  float* dis0 = alloc(N0); float* dis1 = alloc(N1); float* dis2 = alloc(N2);
  alloc(4);                                 // keep 16B alignment downstream
  float* BIG0 = alloc((size_t)N0 * HDIM);   // h0 (skip, kept to the end)
  float* BIG1 = alloc((size_t)N0 * HDIM);   // xw0, later xu0
  float* P1a = alloc((size_t)N1 * HDIM);    // p1, later xu
  float* P1b = alloc((size_t)N1 * HDIM);    // x1, later xw3
  float* P1c = alloc((size_t)N1 * HDIM);    // xw1, later h3
  float* P1d = alloc((size_t)N1 * HDIM);    // h1 (skip)
  float* Q2a = alloc((size_t)N2 * HDIM);    // p2, later h2
  float* Q2b = alloc((size_t)N2 * HDIM);    // x2
  float* Q2c = alloc((size_t)N2 * HDIM);    // xw2
  float* F4  = alloc((size_t)N0 * CC);      // xw4

  _Float16* harena = (_Float16*)(w + ((off + 7) & ~(size_t)7));
  size_t hoff = 0;
  auto halloc = [&](size_t n) { _Float16* p = harena + hoff; hoff += n; return p; };
  _Float16* tW0  = halloc(HDIM * HDIM);          // [128][128]
  _Float16* tW1  = halloc(HDIM * HDIM);
  _Float16* tW2  = halloc(HDIM * HDIM);
  _Float16* tW3  = halloc((size_t)HDIM * 2 * HDIM);  // [128][256]
  _Float16* tW4  = halloc((size_t)16 * 2 * HDIM);    // [16][256], zero-padded
  _Float16* tpW0 = halloc(HDIM * HDIM);
  _Float16* tpW1 = halloc(HDIM * HDIM);
  _Float16* tpW2 = halloc(HDIM * HDIM);
  _Float16* tpW3 = halloc(HDIM * HDIM);

  auto eb = [](int n) { return (n + 255) / 256; };   // elementwise blocks
  auto wb = [](int n) { return (n + 7) / 8; };       // wave-per-row blocks

  auto cvt = [&](const float* Wp, _Float16* Bt, int K, int N, int Np) {
    convert_wT_k<<<eb(Np * K), 256, 0, stream>>>(Wp, Bt, K, N, Np);
  };
  auto gemm128 = [&](const float* A0p, const int* aidx, const _Float16* Bt,
                     const float* biasp, float* Yp, int M, int N, int Np) {
    int tiles = (M / 16) * (Np / 16);
    wmma_gemm_k<128><<<(tiles + 7) / 8, 256, 0, stream>>>(
        A0p, A0p, 128, aidx, Bt, biasp, Yp, M, N, Np);
  };
  auto gemm256 = [&](const float* A0p, const float* A1p, const _Float16* Bt,
                     const float* biasp, float* Yp, int M, int N, int Np) {
    int tiles = (M / 16) * (Np / 16);
    wmma_gemm_k<256><<<(tiles + 7) / 8, 256, 0, stream>>>(
        A0p, A1p, 128, nullptr, Bt, biasp, Yp, M, N, Np);
  };

  // ---- one-time weight convert+transpose (f16, padded)
  cvt(W0, tW0, HDIM, HDIM, HDIM);
  cvt(W1, tW1, HDIM, HDIM, HDIM);
  cvt(W2, tW2, HDIM, HDIM, HDIM);
  cvt(W3, tW3, 2 * HDIM, HDIM, HDIM);
  cvt(W4, tW4, 2 * HDIM, CC, 16);
  cvt(pW + 0 * HDIM * HDIM, tpW0, HDIM, HDIM, HDIM);
  cvt(pW + 1 * HDIM * HDIM, tpW1, HDIM, HDIM, HDIM);
  cvt(pW + 2 * HDIM * HDIM, tpW2, HDIM, HDIM, HDIM);
  cvt(pW + 3 * HDIM * HDIM, tpW3, HDIM, HDIM, HDIM);

  // ---- degrees -> dis = (1 + indeg)^-1/2 per level
  fill_k<<<eb(N0), 256, 0, stream>>>(dis0, 1.0f, N0);
  deg_count_k<<<eb(E0), 256, 0, stream>>>(ei0 + E0, E0, dis0);
  rsqrt_k<<<eb(N0), 256, 0, stream>>>(dis0, N0);
  fill_k<<<eb(N1), 256, 0, stream>>>(dis1, 1.0f, N1);
  deg_count_k<<<eb(E1), 256, 0, stream>>>(ei1 + E1, E1, dis1);
  rsqrt_k<<<eb(N1), 256, 0, stream>>>(dis1, N1);
  fill_k<<<eb(N2), 256, 0, stream>>>(dis2, 1.0f, N2);
  deg_count_k<<<eb(E2), 256, 0, stream>>>(ei2 + E2, E2, dis2);
  rsqrt_k<<<eb(N2), 256, 0, stream>>>(dis2, N2);

  // ---- level 0 GCN: h0 = relu(agg(x@W0) + b0)
  gemm128(x, nullptr, tW0, nullptr, BIG1, N0, HDIM, HDIM);
  agg_self128<<<wb(N0), 256, 0, stream>>>(BIG1, dis0, BIG0, N0);
  agg_edge128<<<wb(E0), 256, 0, stream>>>(ei0, ei0 + E0, dis0, BIG1, BIG0, E0);
  bias_relu_k<<<eb(N0 * HDIM), 256, 0, stream>>>(BIG0, b0, N0 * HDIM);

  // ---- pool 0: x1 = relu(BN(relu(segsum(h0,par1)@pW0+pb0)))
  fill_k<<<eb(N1 * HDIM), 256, 0, stream>>>(P1a, 0.f, N1 * HDIM);
  scatter_parent128<<<wb(N0), 256, 0, stream>>>(BIG0, par1, P1a, N0);
  gemm128(P1a, nullptr, tpW0, pb, P1b, N1, HDIM, HDIM);
  bn_relu_k<<<eb(N1 * HDIM), 256, 0, stream>>>(P1b, pg, pbt, pmn, pvr, N1 * HDIM);

  // ---- level 1 GCN: h1
  gemm128(P1b, nullptr, tW1, nullptr, P1c, N1, HDIM, HDIM);
  agg_self128<<<wb(N1), 256, 0, stream>>>(P1c, dis1, P1d, N1);
  agg_edge128<<<wb(E1), 256, 0, stream>>>(ei1, ei1 + E1, dis1, P1c, P1d, E1);
  bias_relu_k<<<eb(N1 * HDIM), 256, 0, stream>>>(P1d, b1, N1 * HDIM);

  // ---- pool 1: x2
  fill_k<<<eb(N2 * HDIM), 256, 0, stream>>>(Q2a, 0.f, N2 * HDIM);
  scatter_parent128<<<wb(N1), 256, 0, stream>>>(P1d, par2, Q2a, N1);
  gemm128(Q2a, nullptr, tpW1, pb + HDIM, Q2b, N2, HDIM, HDIM);
  bn_relu_k<<<eb(N2 * HDIM), 256, 0, stream>>>(Q2b, pg + HDIM, pbt + HDIM,
                                               pmn + HDIM, pvr + HDIM, N2 * HDIM);

  // ---- level 2 GCN: h2 (into Q2a; p2 is dead)
  gemm128(Q2b, nullptr, tW2, nullptr, Q2c, N2, HDIM, HDIM);
  agg_self128<<<wb(N2), 256, 0, stream>>>(Q2c, dis2, Q2a, N2);
  agg_edge128<<<wb(E2), 256, 0, stream>>>(ei2, ei2 + E2, dis2, Q2c, Q2a, E2);
  bias_relu_k<<<eb(N2 * HDIM), 256, 0, stream>>>(Q2a, b2, N2 * HDIM);

  // ---- unpool 1: xu = relu(BN(relu(h2[par2]@pW2+pb2)))  (gather fused in GEMM)
  gemm128(Q2a, par2, tpW2, pb + 2 * HDIM, P1a, N1, HDIM, HDIM);
  bn_relu_k<<<eb(N1 * HDIM), 256, 0, stream>>>(P1a, pg + 2 * HDIM, pbt + 2 * HDIM,
                                               pmn + 2 * HDIM, pvr + 2 * HDIM, N1 * HDIM);

  // ---- up GCN 1: h3 = relu(agg(concat(xu,h1)@W3) + b3)  (concat fused in GEMM)
  gemm256(P1a, P1d, tW3, nullptr, P1b, N1, HDIM, HDIM);
  agg_self128<<<wb(N1), 256, 0, stream>>>(P1b, dis1, P1c, N1);
  agg_edge128<<<wb(E1), 256, 0, stream>>>(ei1, ei1 + E1, dis1, P1b, P1c, E1);
  bias_relu_k<<<eb(N1 * HDIM), 256, 0, stream>>>(P1c, b3, N1 * HDIM);

  // ---- unpool 0: xu0 = relu(BN(relu(h3[par1]@pW3+pb3)))
  gemm128(P1c, par1, tpW3, pb + 3 * HDIM, BIG1, N0, HDIM, HDIM);
  bn_relu_k<<<eb(N0 * HDIM), 256, 0, stream>>>(BIG1, pg + 3 * HDIM, pbt + 3 * HDIM,
                                               pmn + 3 * HDIM, pvr + 3 * HDIM, N0 * HDIM);

  // ---- final GCN: out = agg(concat(xu0,h0)@W4) + b4, then log_softmax
  gemm256(BIG1, BIG0, tW4, nullptr, F4, N0, CC, 16);
  agg_self_c<<<eb(N0 * CC), 256, 0, stream>>>(F4, dis0, out, N0, CC);
  agg_edge_c<<<eb(E0 * CC), 256, 0, stream>>>(ei0, ei0 + E0, dis0, F4, out, E0, CC);
  bias_add_c<<<eb(N0 * CC), 256, 0, stream>>>(out, b4, N0 * CC, CC);
  log_softmax_k<<<eb(N0), 256, 0, stream>>>(out, N0, CC);
}
